// Decoder_48979807043670
// MI455X (gfx1250) — matrix operand order repaired and checked by reading, not compiled, not run on hardware
//
#include <hip/hip_runtime.h>
#include <hip/hip_bf16.h>

typedef __bf16 bf16;
typedef __attribute__((ext_vector_type(16))) __bf16 v16bf;
typedef __attribute__((ext_vector_type(8)))  float  v8f;

#define N_NODES 1000
#define BATCH   16
#define SEQ     12
#define HID     64
#define E_EDGES 16000
#define ETOT    17000
#define NB      16000   // N_NODES*BATCH

__device__ __forceinline__ float sigmoidf(float x){ return 1.0f/(1.0f+__expf(-x)); }

// ---- WMMA fragment loaders (contiguous k = 16*(lane/16)+slot, bf16 16x16x32) ----
// A (16xK tile, bf16, LDS, row-major with stride ldst):
//   lane<16 : M=lane,     slots 0..15 = K=kb+0..15
//   lane>=16: M=lane-16,  slots 0..15 = K=kb+16..31
__device__ __forceinline__ v16bf load_A_lds(const bf16* At, int ldst, int lane, int kbase){
  int m     = lane & 15;
  int khalf = (lane & 16) ? 16 : 0;
  const bf16* p = At + m*ldst + kbase + khalf;
  v16bf a;
#pragma unroll
  for (int j=0;j<16;++j) a[j]   = p[j];
  return a;
}
// B (KxN tile). Weights stored as W[n][k] row-major (i.e. B transposed), stride ldk.
//   lane<16 : N=lane,    slots 0..15 = K=kb+0..15
//   lane>=16: N=lane-16, slots 0..15 = K=kb+16..31
__device__ __forceinline__ v16bf load_B_glb(const bf16* W, int ldk, int lane, int nbase, int kbase){
  int n    = lane & 15;
  int koff = (lane & 16) ? 16 : 0;
  const bf16* p = W + (size_t)(nbase + n)*ldk + kbase + koff;
  v16bf b;
#pragma unroll
  for (int j=0;j<16;++j) b[j] = p[j];
  return b;
}

// ---- one-time prep: weights -> bf16 in [n][k] layout; attention matrices from W3/b3 ----
__global__ void wprep_kernel(const float* __restrict__ Wh1, const float* __restrict__ Wi2,
                             const float* __restrict__ Wh2, const float* __restrict__ W3,
                             const float* __restrict__ b3,
                             bf16* __restrict__ Wh1b, bf16* __restrict__ Wi2b,
                             bf16* __restrict__ Wh2b, bf16* __restrict__ Wab){
  int i = blockIdx.x*blockDim.x + threadIdx.x;
  if (i < 12288){                       // 192x64 matrices, already [n][k] row-major
    Wh1b[i] = (bf16)Wh1[i];
    Wi2b[i] = (bf16)Wi2[i];
    Wh2b[i] = (bf16)Wh2[i];
  }
  if (i < 24576){                       // 6 x 64 x 64, [m][n][k]; m:0..2 top(dst), 3..5 bot(src)
    int m = i >> 12, n = (i >> 6) & 63, k = i & 63;
    int k128 = (m < 3) ? k : (64 + k);
    int mm = m % 3;                     // 0->W3[:,0] (a), 1->W3[:,1] (b), 2->b3
    float v = (mm < 2) ? W3[(k128*64 + n)*2 + mm] : b3[k128*64 + n];
    Wab[i] = (bf16)v;
  }
}

// ---- one-time: per-edge MLP coefficients a_e, b_e ----
__global__ void edge_coef_kernel(const int* __restrict__ edge_index, const float* __restrict__ features,
                                 const float* __restrict__ W1, const float* __restrict__ b1,
                                 const float* __restrict__ W2, const float* __restrict__ b2,
                                 float* __restrict__ eab){
  int e = blockIdx.x*blockDim.x + threadIdx.x;
  if (e >= ETOT) return;
  int s, d;
  if (e < E_EDGES){ s = edge_index[e]; d = edge_index[E_EDGES + e]; }
  else            { s = d = e - E_EDGES; }
  const float* fd = features + d*64;
  const float* fs = features + s*64;
  float y[16];
#pragma unroll 1
  for (int j=0;j<16;++j){
    const float* w = W1 + j*128;
    float acc = b1[j];
    for (int k=0;k<64;++k) acc += fd[k]*w[k];
    for (int k=0;k<64;++k) acc += fs[k]*w[64+k];
    y[j] = sigmoidf(acc);
  }
  float a = b2[0], bb = b2[1];
#pragma unroll
  for (int j=0;j<16;++j){ a += y[j]*W2[j]; bb += y[j]*W2[16+j]; }
  eab[2*e]   = sigmoidf(a);
  eab[2*e+1] = sigmoidf(bb);
}

// ---- one-time: per-node feature part of the output projection ----
__global__ void fdot_kernel(const float* __restrict__ features, const float* __restrict__ Wo,
                            float* __restrict__ fdot){
  int n = blockIdx.x*blockDim.x + threadIdx.x;
  if (n >= N_NODES) return;
  float s = 0.f;
  for (int c=0;c<64;++c) s += features[n*64+c]*Wo[64+c];
  fdot[n] = s;
}

__global__ void zero_kernel(float* __restrict__ p, int n){
  int i = blockIdx.x*blockDim.x + threadIdx.x;
  if (i < n) p[i] = 0.f;
}

// ---- GRU cell 1: h0n = GRU(data, h0); gh via WMMA, gi (K=3) in VALU ----
__global__ void __launch_bounds__(128)
gru1_kernel(const float* __restrict__ h0_in, const float* __restrict__ pred_prev,
            const float* __restrict__ labels, const bf16* __restrict__ Wh1b,
            const float* __restrict__ Wi1, const float* __restrict__ bi1,
            const float* __restrict__ bh1, float* __restrict__ h0_out, int t){
  __shared__ bf16  Ah[16][72];
  __shared__ float G[16][196];
  int tid  = threadIdx.x;
  int row0 = blockIdx.x * 16;
  { int r = tid >> 3, c0 = (tid & 7)*8;
    const float* src = h0_in + (size_t)(row0 + r)*64 + c0;
#pragma unroll
    for (int j=0;j<8;++j) Ah[r][c0+j] = (bf16)src[j];
  }
  __syncthreads();
  int lane = tid & 31, wave = tid >> 5;
#pragma unroll 1
  for (int ti=0; ti<3; ++ti){                 // 4 waves x 3 tiles = 192 cols
    int nbase = (wave*3 + ti)*16;
    v8f acc = {};
    v16bf a0 = load_A_lds(&Ah[0][0], 72, lane, 0);
    v16bf b0 = load_B_glb(Wh1b, 64, lane, nbase, 0);
    acc = __builtin_amdgcn_wmma_f32_16x16x32_bf16(false, a0, false, b0, (short)0, acc, false, false);
    v16bf a1 = load_A_lds(&Ah[0][0], 72, lane, 32);
    v16bf b1 = load_B_glb(Wh1b, 64, lane, nbase, 32);
    acc = __builtin_amdgcn_wmma_f32_16x16x32_bf16(false, a1, false, b1, (short)0, acc, false, false);
    int nn = lane & 15, mbase = (lane & 16) ? 8 : 0;
#pragma unroll
    for (int v=0; v<8; ++v) G[mbase+v][nbase+nn] = acc[v];
  }
  __syncthreads();
#pragma unroll 1
  for (int i=0;i<8;++i){
    int idx = tid*8 + i;
    int r = idx >> 6, c = idx & 63;
    int row = row0 + r;
    float d0=0.f, d1=0.f, d2=0.f;
    if (t > 0){
      d0 = pred_prev[row];
      size_t base = ((size_t)row*SEQ + (t-1))*3;      // labels (N,B,L,3), row = n*B+b
      d1 = labels[base+1];
      d2 = labels[base+2];
    }
    float gi_r = bi1[c]     + d0*Wi1[c*3]       + d1*Wi1[c*3+1]       + d2*Wi1[c*3+2];
    float gi_z = bi1[c+64]  + d0*Wi1[(c+64)*3]  + d1*Wi1[(c+64)*3+1]  + d2*Wi1[(c+64)*3+2];
    float gi_n = bi1[c+128] + d0*Wi1[(c+128)*3] + d1*Wi1[(c+128)*3+1] + d2*Wi1[(c+128)*3+2];
    float gh_r = G[r][c]     + bh1[c];
    float gh_z = G[r][c+64]  + bh1[c+64];
    float gh_n = G[r][c+128] + bh1[c+128];
    float rg = sigmoidf(gi_r + gh_r);
    float zg = sigmoidf(gi_z + gh_z);
    float ng = tanhf(gi_n + rg*gh_n);
    float h  = h0_in[(size_t)row*64 + c];
    h0_out[(size_t)row*64 + c] = (1.f - zg)*ng + zg*h;
  }
}

// ---- attention node projections: PQ[row][m*64+c], 6 matrices, all WMMA ----
__global__ void __launch_bounds__(128)
attn_proj_kernel(const float* __restrict__ h0n, const bf16* __restrict__ Wab,
                 float* __restrict__ PQ){
  __shared__ bf16 Ah[16][72];
  int tid  = threadIdx.x;
  int row0 = blockIdx.x*16;
  { int r = tid >> 3, c0 = (tid & 7)*8;
    const float* src = h0n + (size_t)(row0 + r)*64 + c0;
#pragma unroll
    for (int j=0;j<8;++j) Ah[r][c0+j] = (bf16)src[j];
  }
  __syncthreads();
  int lane = tid & 31, wave = tid >> 5;
#pragma unroll 1
  for (int i=0;i<6;++i){                      // 4 waves x 6 = 24 tiles (6 mats x 4 ntiles)
    int tt = wave*6 + i;
    int m = tt >> 2, nbase = (tt & 3)*16;
    const bf16* Bm = Wab + m*4096;
    v8f acc = {};
    acc = __builtin_amdgcn_wmma_f32_16x16x32_bf16(false, load_A_lds(&Ah[0][0],72,lane,0),
                                                  false, load_B_glb(Bm,64,lane,nbase,0),
                                                  (short)0, acc, false, false);
    acc = __builtin_amdgcn_wmma_f32_16x16x32_bf16(false, load_A_lds(&Ah[0][0],72,lane,32),
                                                  false, load_B_glb(Bm,64,lane,nbase,32),
                                                  (short)0, acc, false, false);
    int nn = lane & 15, mbase = (lane & 16) ? 8 : 0;
#pragma unroll
    for (int v=0; v<8; ++v)
      PQ[(size_t)(row0+mbase+v)*384 + m*64 + nbase + nn] = acc[v];
  }
}

// ---- per-edge: alpha, softmax over batch axis, segment-sum via f32 atomics ----
__global__ void __launch_bounds__(256)
edge_attn_kernel(const int* __restrict__ edge_index, const float* __restrict__ eab,
                 const float* __restrict__ PQ, const float* __restrict__ h0n,
                 float* __restrict__ agg){
  int tid = threadIdx.x;
  int e = blockIdx.x*4 + (tid >> 6);
  int c = tid & 63;
  if (e >= ETOT) return;
  int s, d;
  if (e < E_EDGES){ s = edge_index[e]; d = edge_index[E_EDGES + e]; }
  else            { s = d = e - E_EDGES; }
  float a = eab[2*e], bco = eab[2*e+1];
  float al[16];
  float vmax = -1e30f;
#pragma unroll
  for (int b=0;b<16;++b){
    const float* Pd = PQ + (size_t)(d*16 + b)*384;
    const float* Ps = PQ + (size_t)(s*16 + b)*384;
    float x = a*(Pd[c] + Ps[192+c]) + bco*(Pd[64+c] + Ps[256+c]) + Pd[128+c] + Ps[320+c];
    x = (x > 0.f) ? x : 0.01f*x;                 // leaky_relu(0.01)
    al[b] = x;
    vmax = fmaxf(vmax, x);
  }
  float ssum = 0.f;
#pragma unroll
  for (int b=0;b<16;++b){ al[b] = __expf(al[b]-vmax); ssum += al[b]; }
  float inv = 1.f/ssum;
#pragma unroll
  for (int b=0;b<16;++b){
    float w = al[b]*inv;
    float contrib = w * h0n[(size_t)(s*16 + b)*64 + c];
    atomicAdd(&agg[(size_t)(d*16 + b)*64 + c], contrib);
  }
}

// ---- GRU cell 2 + output projection; both GEMMs via WMMA ----
__global__ void __launch_bounds__(128)
gru2_kernel(const float* __restrict__ agg, const float* __restrict__ h1_in,
            const bf16* __restrict__ Wi2b, const bf16* __restrict__ Wh2b,
            const float* __restrict__ bi2, const float* __restrict__ bh2,
            const float* __restrict__ Wo, const float* __restrict__ bo,
            const float* __restrict__ fdot, float* __restrict__ h1_out,
            float* __restrict__ pred_out, float* __restrict__ out, int t){
  __shared__ bf16  Ag[16][72];
  __shared__ bf16  Ahh[16][72];
  __shared__ float Gi[16][196];
  __shared__ float Gh[16][196];
  __shared__ float Hn[16][68];
  int tid  = threadIdx.x;
  int row0 = blockIdx.x*16;
  { int r = tid >> 3, c0 = (tid & 7)*8;
    const float* sa = agg   + (size_t)(row0 + r)*64 + c0;
    const float* sh = h1_in + (size_t)(row0 + r)*64 + c0;
#pragma unroll
    for (int j=0;j<8;++j){
      float v = sa[j];
      Ag[r][c0+j]  = (bf16)(v > 0.f ? v : 0.f);  // relu(agg) = gat
      Ahh[r][c0+j] = (bf16)sh[j];
    }
  }
  __syncthreads();
  int lane = tid & 31, wave = tid >> 5;
#pragma unroll 1
  for (int i=0;i<6;++i){                      // 24 tiles: 12 gi + 12 gh
    int tt = wave*6 + i;
    bool is_gi = tt < 12;
    int nbase = (is_gi ? tt : tt-12)*16;
    const bf16* Bw = is_gi ? Wi2b : Wh2b;
    const bf16* At = is_gi ? &Ag[0][0] : &Ahh[0][0];
    v8f acc = {};
    acc = __builtin_amdgcn_wmma_f32_16x16x32_bf16(false, load_A_lds(At,72,lane,0),
                                                  false, load_B_glb(Bw,64,lane,nbase,0),
                                                  (short)0, acc, false, false);
    acc = __builtin_amdgcn_wmma_f32_16x16x32_bf16(false, load_A_lds(At,72,lane,32),
                                                  false, load_B_glb(Bw,64,lane,nbase,32),
                                                  (short)0, acc, false, false);
    float* Gd = is_gi ? &Gi[0][0] : &Gh[0][0];
    int nn = lane & 15, mbase = (lane & 16) ? 8 : 0;
#pragma unroll
    for (int v=0; v<8; ++v) Gd[(mbase+v)*196 + nbase + nn] = acc[v];
  }
  __syncthreads();
#pragma unroll 1
  for (int i=0;i<8;++i){
    int idx = tid*8 + i;
    int r = idx >> 6, c = idx & 63;
    int row = row0 + r;
    float gr = sigmoidf(Gi[r][c]     + bi2[c]     + Gh[r][c]     + bh2[c]);
    float gz = sigmoidf(Gi[r][c+64]  + bi2[c+64]  + Gh[r][c+64]  + bh2[c+64]);
    float gn = tanhf  (Gi[r][c+128] + bi2[c+128] + gr*(Gh[r][c+128] + bh2[c+128]));
    float h  = h1_in[(size_t)row*64 + c];
    float hn = (1.f - gz)*gn + gz*h;
    h1_out[(size_t)row*64 + c] = hn;
    Hn[r][c] = hn * Wo[c];                    // pre-multiplied for reduction
  }
  __syncthreads();
  if (tid < 16){
    int r = tid, row = row0 + r, n = row >> 4;
    float sacc = bo[0] + fdot[n];
    for (int c=0;c<64;++c) sacc += Hn[r][c];
    pred_out[row] = sacc;
    out[(size_t)row*SEQ + t] = sacc;          // out (N,B,L,1) -> row*L + t
  }
}

extern "C" void kernel_launch(void* const* d_in, const int* in_sizes, int n_in,
                              void* d_out, int out_size, void* d_ws, size_t ws_size,
                              hipStream_t stream){
  (void)in_sizes; (void)n_in; (void)out_size; (void)ws_size;
  const float* hidden0    = (const float*)d_in[0];
  const float* hidden1    = (const float*)d_in[1];
  const float* features   = (const float*)d_in[2];
  const float* labels     = (const float*)d_in[3];
  const int*   edge_index = (const int*)  d_in[4];
  const float* Wi1 = (const float*)d_in[5];
  const float* Wh1 = (const float*)d_in[6];
  const float* bi1 = (const float*)d_in[7];
  const float* bh1 = (const float*)d_in[8];
  const float* W1  = (const float*)d_in[9];
  const float* b1  = (const float*)d_in[10];
  const float* W2  = (const float*)d_in[11];
  const float* b2  = (const float*)d_in[12];
  const float* W3  = (const float*)d_in[13];
  const float* b3  = (const float*)d_in[14];
  const float* Wi2 = (const float*)d_in[15];
  const float* Wh2 = (const float*)d_in[16];
  const float* bi2 = (const float*)d_in[17];
  const float* bh2 = (const float*)d_in[18];
  const float* Wo  = (const float*)d_in[19];
  const float* bo  = (const float*)d_in[20];
  float* out = (float*)d_out;

  char* ws = (char*)d_ws;
  size_t off = 0;
  auto take = [&](size_t bytes)->char*{
    char* p = ws + off;
    off = (off + bytes + 255) & ~(size_t)255;
    return p;
  };
  float* h0buf[2]   = { (float*)take((size_t)NB*64*4), (float*)take((size_t)NB*64*4) };
  float* h1buf[2]   = { (float*)take((size_t)NB*64*4), (float*)take((size_t)NB*64*4) };
  float* PQ         =   (float*)take((size_t)NB*384*4);
  float* agg        =   (float*)take((size_t)NB*64*4);
  float* predbuf[2] = { (float*)take((size_t)NB*4), (float*)take((size_t)NB*4) };
  float* eab        =   (float*)take((size_t)ETOT*2*4);
  float* fdot       =   (float*)take((size_t)N_NODES*4);
  bf16*  Wh1b       =   (bf16*) take(12288*2);
  bf16*  Wi2b       =   (bf16*) take(12288*2);
  bf16*  Wh2b       =   (bf16*) take(12288*2);
  bf16*  Wab        =   (bf16*) take(24576*2);

  // initial states (never mutate inputs)
  hipMemcpyAsync(h0buf[0], hidden0, (size_t)NB*64*4, hipMemcpyDeviceToDevice, stream);
  hipMemcpyAsync(h1buf[0], hidden1, (size_t)NB*64*4, hipMemcpyDeviceToDevice, stream);

  wprep_kernel<<<96, 256, 0, stream>>>(Wh1, Wi2, Wh2, W3, b3, Wh1b, Wi2b, Wh2b, Wab);
  edge_coef_kernel<<<(ETOT+255)/256, 256, 0, stream>>>(edge_index, features, W1, b1, W2, b2, eab);
  fdot_kernel<<<(N_NODES+255)/256, 256, 0, stream>>>(features, Wo, fdot);

  for (int t = 0; t < SEQ; ++t){
    int cur = t & 1, nxt = (t + 1) & 1;
    gru1_kernel<<<NB/16, 128, 0, stream>>>(h0buf[cur], predbuf[cur ^ 1], labels, Wh1b,
                                           Wi1, bi1, bh1, h0buf[nxt], t);
    attn_proj_kernel<<<NB/16, 128, 0, stream>>>(h0buf[nxt], Wab, PQ);
    zero_kernel<<<(NB*64+255)/256, 256, 0, stream>>>(agg, NB*64);
    edge_attn_kernel<<<(ETOT+3)/4, 256, 0, stream>>>(edge_index, eab, PQ, h0buf[nxt], agg);
    gru2_kernel<<<NB/16, 128, 0, stream>>>(agg, h1buf[cur], Wi2b, Wh2b, bi2, bh2,
                                           Wo, bo, fdot, h1buf[nxt], predbuf[cur], out, t);
  }
}
